// Router_67491116089709
// MI455X (gfx1250) — compile-verified, matrix operand order checked by reading, not run on hardware
//
#include <hip/hip_runtime.h>
#include <hip/hip_bf16.h>
#include <math.h>

#define TOKENS  131072
#define HIDDEN  2048
#define EXPERTS 64
#define TOPK    8

#define ET (EXPERTS/16)        // 4 expert tiles of 16
#define KC (HIDDEN/32)         // 64 K-chunks of 32
#define FRAG_ELEMS 16          // bf16 elems per lane per fragment
#define REGION_ELEMS ((size_t)ET*KC*32*FRAG_ELEMS)  // 131072 bf16 per region (= 64x2048)

#define WAVES   8
#define A_TILES 4              // 64 tokens per wave
#define WG_TOKENS (WAVES*64)   // 512 tokens per workgroup
#define ROWSTRIDE 65           // padded LDS row (bank-conflict avoidance)

typedef __attribute__((ext_vector_type(16))) __bf16 v16bf;
typedef __attribute__((ext_vector_type(8)))  float  v8f;
typedef __attribute__((ext_vector_type(4)))  float  f32x4;   // native vector (NT-load OK)
typedef __attribute__((ext_vector_type(4)))  unsigned int u32x4;

union BFrag {
    v16bf          bf;
    unsigned short us[16];
    u32x4          q[2];
};

// ---------------------------------------------------------------------------
// Kernel 1: repack gate_w (f32) into bf16 hi/lo WMMA-B fragments in d_ws.
// B fragment (32x16 bf16, 16x16x32 WMMA): lane = n + 16*khi (n = expert col),
// half h -> k = h + (h>=8 ? 8 : 0) + khi*8 within the 32-wide K chunk.
// hi = truncate-to-bf16(w); lo = bf16(w - hi)  (residual is exact in f32).
// ---------------------------------------------------------------------------
__global__ void gate_pack_kernel(const float* __restrict__ gate_w,
                                 unsigned short* __restrict__ ws) {
    int tid  = blockIdx.x * blockDim.x + threadIdx.x;   // 0..8191
    int lane = tid & 31;
    int frag = tid >> 5;                                // et*KC + kc
    int kc   = frag & (KC - 1);
    int et   = frag >> 6;
    int n    = lane & 15;
    int khi  = lane >> 4;
    int e    = et * 16 + n;
    const float* row = gate_w + (size_t)e * HIDDEN;
    int k0 = kc * 32 + khi * 8;

    BFrag hi, lo;
#pragma unroll
    for (int h = 0; h < 16; ++h) {
        int k = k0 + (h < 8 ? h : 8 + h);               // h>=8 -> +16 + (h-8)
        float g = row[k];
        unsigned u = __float_as_uint(g);
        unsigned short hu = (unsigned short)(u >> 16);  // truncate -> bf16 hi
        float hf = __uint_as_float((unsigned)hu << 16);
        float lf = g - hf;                              // exact residual
        hi.us[h] = hu;
        lo.us[h] = (unsigned short)(__float_as_uint(lf) >> 16);
    }
    size_t off = ((size_t)frag * 32 + lane) * FRAG_ELEMS;
    u32x4* dhi = (u32x4*)(ws + off);
    u32x4* dlo = (u32x4*)(ws + REGION_ELEMS + off);
    dhi[0] = hi.q[0]; dhi[1] = hi.q[1];
    dlo[0] = lo.q[0]; dlo[1] = lo.q[1];
}

// ---------------------------------------------------------------------------
// Kernel 2: router. Each wave: 64 tokens x 64 experts via bf16x2-split WMMA
// (3 products: a_lo*b_hi + a_hi*b_lo + a_hi*b_hi -> ~f32-accurate logits).
// Then sigmoid -> LDS -> per-token biased top-8 + renormalize.
// ---------------------------------------------------------------------------
__global__ __launch_bounds__(256)
void router_topk_kernel(const float* __restrict__ x,
                        const unsigned short* __restrict__ ws,
                        const float* __restrict__ bias,
                        float* __restrict__ out) {
    __shared__ float s_logits[WG_TOKENS * ROWSTRIDE];
    __shared__ float s_bias[EXPERTS];

    const int tid  = threadIdx.x;
    const int lane = tid & 31;
    const int wv   = tid >> 5;
    const int n    = lane & 15;
    const int khi  = lane >> 4;
    const int tokBase = blockIdx.x * WG_TOKENS + wv * 64;

    if (tid < EXPERTS) s_bias[tid] = bias[tid];

    v8f zero = {0.f, 0.f, 0.f, 0.f, 0.f, 0.f, 0.f, 0.f};
    v8f acc[A_TILES][ET];
#pragma unroll
    for (int a = 0; a < A_TILES; ++a)
#pragma unroll
        for (int e = 0; e < ET; ++e) acc[a][e] = zero;

    const float* xrow[A_TILES];
#pragma unroll
    for (int a = 0; a < A_TILES; ++a)
        xrow[a] = x + (size_t)(tokBase + a * 16 + n) * HIDDEN;

    const unsigned short* wsl = ws + REGION_ELEMS;

    for (int kc = 0; kc < KC; ++kc) {
        // --- B fragments (bf16 hi/lo) from L2-resident repacked gate_w ---
        BFrag bhi[ET], blo[ET];
#pragma unroll
        for (int e = 0; e < ET; ++e) {
            size_t off = (((size_t)e * KC + kc) * 32 + lane) * FRAG_ELEMS;
            const u32x4* ph = (const u32x4*)(ws  + off);
            const u32x4* pl = (const u32x4*)(wsl + off);
            bhi[e].q[0] = ph[0]; bhi[e].q[1] = ph[1];
            blo[e].q[0] = pl[0]; blo[e].q[1] = pl[1];
        }

        const int k0 = kc * 32 + khi * 8;
#pragma unroll
        for (int a = 0; a < A_TILES; ++a) {
            // --- stream A tile (non-temporal: x read exactly once) ---
            const f32x4* p0 = (const f32x4*)(xrow[a] + k0);
            const f32x4* p1 = (const f32x4*)(xrow[a] + k0 + 16);
            f32x4 v0 = __builtin_nontemporal_load(p0);
            f32x4 v1 = __builtin_nontemporal_load(p0 + 1);
            f32x4 v2 = __builtin_nontemporal_load(p1);
            f32x4 v3 = __builtin_nontemporal_load(p1 + 1);
            float f[16] = {v0.x, v0.y, v0.z, v0.w, v1.x, v1.y, v1.z, v1.w,
                           v2.x, v2.y, v2.z, v2.w, v3.x, v3.y, v3.z, v3.w};
            BFrag ahi, alo;
#pragma unroll
            for (int h = 0; h < 16; ++h) {
                unsigned u = __float_as_uint(f[h]);
                unsigned short hu = (unsigned short)(u >> 16);
                ahi.us[h] = hu;
                float lf = f[h] - __uint_as_float((unsigned)hu << 16);
                alo.us[h] = (unsigned short)(__float_as_uint(lf) >> 16);
            }
            // --- 3-product split GEMM: near-f32 accuracy on bf16 pipes ---
#pragma unroll
            for (int e = 0; e < ET; ++e) {
                acc[a][e] = __builtin_amdgcn_wmma_f32_16x16x32_bf16(
                    false, alo.bf, false, bhi[e].bf, (short)0, acc[a][e], false, false);
                acc[a][e] = __builtin_amdgcn_wmma_f32_16x16x32_bf16(
                    false, ahi.bf, false, blo[e].bf, (short)0, acc[a][e], false, false);
                acc[a][e] = __builtin_amdgcn_wmma_f32_16x16x32_bf16(
                    false, ahi.bf, false, bhi[e].bf, (short)0, acc[a][e], false, false);
            }
        }
    }

    // --- sigmoid + stage logits in LDS (C layout: lane n=expert, row r=token) ---
#pragma unroll
    for (int a = 0; a < A_TILES; ++a) {
#pragma unroll
        for (int e = 0; e < ET; ++e) {
#pragma unroll
            for (int r = 0; r < 8; ++r) {
                float z  = acc[a][e][r];
                float sg = 1.0f / (1.0f + __expf(-z));
                int   tl = wv * 64 + a * 16 + r + khi * 8;
                s_logits[tl * ROWSTRIDE + e * 16 + n] = sg;
            }
        }
    }
    __syncthreads();

    // --- biased top-8 + renormalize: 2 tokens per thread, register-resident
    //     insertion network (no dynamic indexing -> no scratch) ---
#pragma unroll
    for (int t2 = 0; t2 < 2; ++t2) {
        int  tl = tid * 2 + t2;
        long gt = (long)blockIdx.x * WG_TOKENS + tl;
        const float* lg = &s_logits[tl * ROWSTRIDE];

        float bsel[TOPK], bval[TOPK];
        int   bidx[TOPK];
#pragma unroll
        for (int j = 0; j < TOPK; ++j) { bsel[j] = -1e30f; bval[j] = 0.f; bidx[j] = 0; }

        for (int e = 0; e < EXPERTS; ++e) {
            float cv = lg[e];
            float cs = cv + s_bias[e];
            int   ci = e;
#pragma unroll
            for (int j = 0; j < TOPK; ++j) {
                bool  g  = cs > bsel[j];       // strict > : ties keep lower index first
                float ts = g ? bsel[j] : cs;
                float tv = g ? bval[j] : cv;
                int   ti = g ? bidx[j] : ci;
                bsel[j] = g ? cs : bsel[j];
                bval[j] = g ? cv : bval[j];
                bidx[j] = g ? ci : bidx[j];
                cs = ts; cv = tv; ci = ti;
            }
        }

        float sum = 0.f;
#pragma unroll
        for (int j = 0; j < TOPK; ++j) sum += bval[j];
        float inv = 1.0f / sum;
#pragma unroll
        for (int j = 0; j < TOPK; ++j) {
            out[gt * TOPK + j] = bval[j] * inv;                       // weights (f32)
            out[(size_t)TOKENS * TOPK + gt * TOPK + j] = (float)bidx[j]; // indices
        }
    }
}

// ---------------------------------------------------------------------------
extern "C" void kernel_launch(void* const* d_in, const int* in_sizes, int n_in,
                              void* d_out, int out_size, void* d_ws, size_t ws_size,
                              hipStream_t stream) {
    const float* x      = (const float*)d_in[0];
    const float* gate_w = (const float*)d_in[1];
    const float* bias   = (const float*)d_in[2];
    float* out          = (float*)d_out;
    unsigned short* ws  = (unsigned short*)d_ws;   // needs 2*256KB = 512KB scratch

    // 1) repack gate_w into bf16 hi/lo WMMA fragments (tiny, L2-resident after)
    gate_pack_kernel<<<(ET * KC * 32) / 256, 256, 0, stream>>>(gate_w, ws);

    // 2) fused GEMM + sigmoid + biased top-8 + renormalize
    router_topk_kernel<<<TOKENS / WG_TOKENS, 256, 0, stream>>>(x, ws, bias, out);
}
